// GAT22ReLU_53197464928894
// MI455X (gfx1250) — matrix operand order repaired and checked by reading, not compiled, not run on hardware
//
#include <hip/hip_runtime.h>

#define DD 128   // feature dim (== GEMM K)
#define HH 256   // hidden dim of MLP head (2*DD)
#define LL 4     // GAT layers

typedef __attribute__((ext_vector_type(16))) __bf16 v16bf;
typedef __attribute__((ext_vector_type(8)))  float  v8f;

#define CDIV(a,b) (((a)+(b)-1)/(b))

// ---------- helpers ----------
__device__ __forceinline__ unsigned short f2bf(float f) {
  unsigned u = __float_as_uint(f);
  unsigned r = u + 0x7FFFu + ((u >> 16) & 1u);   // round-to-nearest-even
  return (unsigned short)(r >> 16);
}
// monotonic float<->uint map for atomicMax on floats
__device__ __forceinline__ unsigned fmap(float f) {
  unsigned u = __float_as_uint(f);
  return (u & 0x80000000u) ? ~u : (u | 0x80000000u);
}
__device__ __forceinline__ float funmap(unsigned u) {
  unsigned b = (u & 0x80000000u) ? (u & 0x7FFFFFFFu) : ~u;
  return __uint_as_float(b);
}

// ---------- generic utility kernels ----------
__global__ void k_zero(float* p, int n) {
  int i = blockIdx.x * blockDim.x + threadIdx.x;
  if (i < n) p[i] = 0.0f;
}
__global__ void k_cast_bf16(const float* __restrict__ src, unsigned short* __restrict__ dst, int n) {
  int i = blockIdx.x * blockDim.x + threadIdx.x;
  if (i < n) dst[i] = f2bf(src[i]);
}

// ---------- pack f32 weights [K=128, Ncol] into WMMA B-fragment order (bf16) ----------
// Packed index: ((tileN*4 + kk)*32 + lane)*8 + j  holds K = kk*32 + 16*(lane>>4) + 2j (+1 in hi),
// column = tileN*16 + (lane&15).  One thread per packed dword; total = 64*Ncol dwords.
__global__ void k_pack_b(const float* __restrict__ B, unsigned* __restrict__ Bp, int Ncol) {
  int i = blockIdx.x * blockDim.x + threadIdx.x;
  int total = Ncol * 64;
  if (i >= total) return;
  int j     = i & 7;
  int lane  = (i >> 3) & 31;
  int kk    = (i >> 8) & 3;
  int tileN = i >> 10;
  int k0  = kk * 32 + ((lane >> 4) << 4) + 2 * j;
  int col = tileN * 16 + (lane & 15);
  unsigned lo = f2bf(B[(size_t)k0 * Ncol + col]);
  unsigned hi = f2bf(B[(size_t)(k0 + 1) * Ncol + col]);
  Bp[i] = lo | (hi << 16);
}

// ---------- WMMA GEMM: C[M,Ncol] = A[M,128] * B[128,Ncol], bf16 in / f32 out ----------
// One wave computes a 16x128 strip (8 accumulator tiles), reusing the A fragment across
// all 8 WMMAs per k-step. K = 128, fully unrolled -> 32 v_wmma per wave.
// Ncol must be a multiple of 128; M a multiple of 16.
__global__ void k_gemm_bf16_wmma(const unsigned short* __restrict__ A,
                                 const unsigned* __restrict__ Bp,   // packed fragments
                                 float* __restrict__ C,
                                 int M, int Ncol) {
  int wave = (blockIdx.x * blockDim.x + threadIdx.x) >> 5;
  int lane = threadIdx.x & 31;
  int nchunks = Ncol >> 7;                 // 8 column tiles per chunk
  int tileM = wave / nchunks;
  int chunk = wave - tileM * nchunks;
  if (tileM * 16 >= M) return;             // wave-uniform guard (EXEC stays all-ones)
  int half = lane >> 4;
  int l16  = lane & 15;

  v8f acc[8] = {};
  const unsigned short* arow = A + (size_t)(tileM * 16 + l16) * DD + (half << 4);
  const unsigned* bbase = Bp + ((size_t)chunk * 8 * 4 * 32 + lane) * 8;

#pragma unroll
  for (int kk = 0; kk < 4; ++kk) {
    union { v16bf v; unsigned u[8]; } a;
#pragma unroll
    for (int j = 0; j < 8; ++j) {
      int kA = kk * 32 + 2 * j;
      a.u[j] = *(const unsigned*)(arow + kA);
    }
#pragma unroll
    for (int t = 0; t < 8; ++t) {
      union { v16bf v; unsigned u[8]; } b;
      const unsigned* bp = bbase + ((size_t)t * 4 + kk) * 32 * 8;
#pragma unroll
      for (int j = 0; j < 8; ++j) b.u[j] = bp[j];
      acc[t] = __builtin_amdgcn_wmma_f32_16x16x32_bf16(false, a.v, false, b.v,
                                                       (short)0, acc[t], false, false);
    }
  }
  // C layout: VGPR r, lane -> row = r + 8*half, col = l16 (within tile)
#pragma unroll
  for (int t = 0; t < 8; ++t) {
    float* crow = C + (size_t)(tileM * 16 + (half << 3)) * Ncol + (chunk * 8 + t) * 16 + l16;
#pragma unroll
    for (int r = 0; r < 8; ++r) crow[(size_t)r * Ncol] = acc[t][r];
  }
}

// ---------- attention scores: as[n] = <h[n], a_s>, ad[n] = <h[n], a_d> ----------
__global__ void k_attn_scores(const float* __restrict__ h,
                              const float* __restrict__ a_s,
                              const float* __restrict__ a_d,
                              float* __restrict__ as_n, float* __restrict__ ad_n, int N) {
  int gid = blockIdx.x * blockDim.x + threadIdx.x;
  int node = gid >> 5, lane = gid & 31;
  if (node >= N) return;
  const float* row = h + (size_t)node * DD;
  float sa = 0.f, sd = 0.f;
#pragma unroll
  for (int j = 0; j < DD; j += 32) {
    float v = row[lane + j];
    sa += v * a_s[lane + j];
    sd += v * a_d[lane + j];
  }
#pragma unroll
  for (int o = 16; o; o >>= 1) { sa += __shfl_xor(sa, o, 32); sd += __shfl_xor(sd, o, 32); }
  if (lane == 0) { as_n[node] = sa; ad_n[node] = sd; }
}

// ---------- edge kernels ----------
__global__ void k_edge_logit_max(const int* __restrict__ src, const int* __restrict__ dst,
                                 const float* __restrict__ as_n, const float* __restrict__ ad_n,
                                 float* __restrict__ logit, unsigned* __restrict__ m_u, int E) {
  int e = blockIdx.x * blockDim.x + threadIdx.x;
  if (e >= E) return;
  float l = as_n[src[e]] + ad_n[dst[e]];
  l = (l > 0.f) ? l : 0.2f * l;                   // leaky_relu(0.2)
  logit[e] = l;
  atomicMax(&m_u[dst[e]], fmap(l));
}
__global__ void k_edge_exp_sum(const int* __restrict__ dst,
                               const float* __restrict__ logit, const unsigned* __restrict__ m_u,
                               float* __restrict__ eexp, float* __restrict__ denom, int E) {
  int e = blockIdx.x * blockDim.x + threadIdx.x;
  if (e >= E) return;
  int d = dst[e];
  float ev = __expf(logit[e] - funmap(m_u[d]));
  eexp[e] = ev;
  atomicAdd(&denom[d], ev);
}
__global__ void k_edge_aggregate(const int* __restrict__ src, const int* __restrict__ dst,
                                 const float* __restrict__ eexp, const float* __restrict__ denom,
                                 const float* __restrict__ h, float* __restrict__ agg, int E) {
  int tid = blockIdx.x * blockDim.x + threadIdx.x;   // E*32 threads, 4 cols each
  if (tid >= E * 32) return;
  int e  = tid >> 5;
  int c0 = (tid & 31) << 2;
  int s = src[e], d = dst[e];
  float alpha = eexp[e] / fmaxf(denom[d], 1e-16f);
  const float4 hv = *(const float4*)(h + (size_t)s * DD + c0);
  float* o = agg + (size_t)d * DD + c0;
  atomicAdd(o + 0, alpha * hv.x);
  atomicAdd(o + 1, alpha * hv.y);
  atomicAdd(o + 2, alpha * hv.z);
  atomicAdd(o + 3, alpha * hv.w);
}

// ---------- BatchNorm ----------
__global__ void k_bn_stats(const float* __restrict__ agg,
                           float* __restrict__ colsum, float* __restrict__ colsq, int N) {
  int col = threadIdx.x;                              // 128 threads/block
  float s = 0.f, q = 0.f;
  for (int row = blockIdx.x; row < N; row += gridDim.x) {
    float v = agg[(size_t)row * DD + col];
    s += v; q += v * v;
  }
  atomicAdd(&colsum[col], s);
  atomicAdd(&colsq[col], q);
}
__global__ void k_bn_finalize(const float* __restrict__ colsum, const float* __restrict__ colsq,
                              const float* __restrict__ gamma, const float* __restrict__ beta,
                              float* __restrict__ scale, float* __restrict__ shift, int N) {
  int c = threadIdx.x;
  float inv = 1.0f / (float)N;
  float mu  = colsum[c] * inv;
  float var = colsq[c] * inv - mu * mu;
  float sc  = gamma[c] * rsqrtf(var + 1e-5f);
  scale[c] = sc;
  shift[c] = beta[c] - mu * sc;
}
__global__ void k_bn_relu_cast(const float* __restrict__ agg,
                               const float* __restrict__ scale, const float* __restrict__ shift,
                               float* __restrict__ xcur, unsigned short* __restrict__ xb, int n) {
  int i = blockIdx.x * blockDim.x + threadIdx.x;
  if (i >= n) return;
  int c = i & (DD - 1);
  float v = fmaxf(agg[i] * scale[c] + shift[c], 0.f);
  xcur[i] = v;
  xb[i] = f2bf(v);
}

// ---------- pooling + head ----------
__global__ void k_pool_cnt(const int* __restrict__ batch, float* __restrict__ cnt, int N) {
  int n = blockIdx.x * blockDim.x + threadIdx.x;
  if (n < N) atomicAdd(&cnt[batch[n]], 1.0f);
}
__global__ void k_pool_sum(const int* __restrict__ batch, const float* __restrict__ x,
                           float* __restrict__ gsum, int N) {
  int tid = blockIdx.x * blockDim.x + threadIdx.x;
  if (tid >= N * 32) return;
  int n = tid >> 5, c0 = (tid & 31) << 2;
  int g = batch[n];
  const float4 v = *(const float4*)(x + (size_t)n * DD + c0);
  float* o = gsum + (size_t)g * DD + c0;
  atomicAdd(o + 0, v.x); atomicAdd(o + 1, v.y);
  atomicAdd(o + 2, v.z); atomicAdd(o + 3, v.w);
}
__global__ void k_pool_div_cast(const float* __restrict__ gsum, const float* __restrict__ cnt,
                                unsigned short* __restrict__ gbf, int n) {
  int i = blockIdx.x * blockDim.x + threadIdx.x;
  if (i >= n) return;
  int g = i >> 7;                                    // /DD
  gbf[i] = f2bf(gsum[i] / fmaxf(cnt[g], 1.0f));
}
__global__ void k_bias_relu(float* __restrict__ hidden, const float* __restrict__ b1, int n) {
  int i = blockIdx.x * blockDim.x + threadIdx.x;
  if (i >= n) return;
  int c = i & (HH - 1);
  hidden[i] = fmaxf(hidden[i] + b1[c], 0.f);
}
__global__ void k_head2(const float* __restrict__ hidden, const float* __restrict__ W2,
                        const float* __restrict__ b2, float* __restrict__ out, int G) {
  int gid = blockIdx.x * blockDim.x + threadIdx.x;
  int g = gid >> 5, lane = gid & 31;
  if (g >= G) return;
  const float* row = hidden + (size_t)g * HH;
  float s = 0.f;
#pragma unroll
  for (int j = 0; j < HH; j += 32) s += row[lane + j] * W2[lane + j];
#pragma unroll
  for (int o = 16; o; o >>= 1) s += __shfl_xor(s, o, 32);
  if (lane == 0) out[g] = s + b2[0];
}

// ---------- driver ----------
extern "C" void kernel_launch(void* const* d_in, const int* in_sizes, int n_in,
                              void* d_out, int out_size, void* d_ws, size_t ws_size,
                              hipStream_t stream) {
  const float* x_in  = (const float*)d_in[0];
  const int*   eidx  = (const int*)d_in[1];
  const int*   batch = (const int*)d_in[3];
  const float* W     = (const float*)d_in[4];
  const float* att_s = (const float*)d_in[5];
  const float* att_d = (const float*)d_in[6];
  const float* gamma = (const float*)d_in[8];
  const float* beta  = (const float*)d_in[9];
  const float* W1    = (const float*)d_in[10];
  const float* b1    = (const float*)d_in[11];
  const float* W2    = (const float*)d_in[12];
  const float* b2    = (const float*)d_in[13];
  float* out = (float*)d_out;

  const int N = in_sizes[0] / DD;
  const int E = in_sizes[1] / 2;
  const int G = out_size;
  const int* src = eidx;
  const int* dst = eidx + E;

  char* ws = (char*)d_ws;
  size_t off = 0;
  auto alloc = [&](size_t bytes) -> char* {
    char* p = ws + off;
    off += (bytes + 255) & ~(size_t)255;
    return p;
  };
  float* h            = (float*)alloc((size_t)N * DD * 4);
  float* agg          = (float*)alloc((size_t)N * DD * 4);
  float* xcur         = (float*)alloc((size_t)N * DD * 4);
  unsigned short* xb  = (unsigned short*)alloc((size_t)N * DD * 2);
  unsigned* wbp       = (unsigned*)alloc((size_t)HH * 64 * 4);   // packed B fragments
  float* as_n         = (float*)alloc((size_t)N * 4);
  float* ad_n         = (float*)alloc((size_t)N * 4);
  unsigned* m_u       = (unsigned*)alloc((size_t)N * 4);
  float* denom        = (float*)alloc((size_t)N * 4);
  float* logit        = (float*)alloc((size_t)E * 4);
  float* eexp         = (float*)alloc((size_t)E * 4);
  float* colsum       = (float*)alloc(DD * 4);
  float* colsq        = (float*)alloc(DD * 4);
  float* scale        = (float*)alloc(DD * 4);
  float* shift        = (float*)alloc(DD * 4);
  float* gsum         = (float*)alloc((size_t)G * DD * 4);
  float* cnt          = (float*)alloc((size_t)G * 4);
  unsigned short* gbf = (unsigned short*)alloc((size_t)G * DD * 2);
  float* hidden       = (float*)alloc((size_t)G * HH * 4);
  (void)ws_size; (void)n_in;

  // initial activation cast to bf16
  k_cast_bf16<<<CDIV(N * DD, 256), 256, 0, stream>>>(x_in, xb, N * DD);

  for (int l = 0; l < LL; ++l) {
    // h = x @ W[l]  (WMMA bf16 -> f32), weights pre-packed into fragment layout
    k_pack_b<<<CDIV(DD * 64, 256), 256, 0, stream>>>(W + (size_t)l * DD * DD, wbp, DD);
    int waves = (N / 16) * (DD / 128);
    k_gemm_bf16_wmma<<<CDIV(waves, 8), 256, 0, stream>>>(xb, wbp, h, N, DD);

    // attention scores
    k_attn_scores<<<CDIV(N * 32, 256), 256, 0, stream>>>(h, att_s + (size_t)l * DD,
                                                         att_d + (size_t)l * DD, as_n, ad_n, N);
    // segment softmax + aggregation
    k_zero<<<CDIV(N, 256), 256, 0, stream>>>((float*)m_u, N);
    k_zero<<<CDIV(N, 256), 256, 0, stream>>>(denom, N);
    k_zero<<<CDIV(N * DD, 256), 256, 0, stream>>>(agg, N * DD);
    k_edge_logit_max<<<CDIV(E, 256), 256, 0, stream>>>(src, dst, as_n, ad_n, logit, m_u, E);
    k_edge_exp_sum<<<CDIV(E, 256), 256, 0, stream>>>(dst, logit, m_u, eexp, denom, E);
    k_edge_aggregate<<<CDIV(E * 32, 256), 256, 0, stream>>>(src, dst, eexp, denom, h, agg, E);

    // BatchNorm (bias cancels inside BN) + ReLU + bf16 cast for next GEMM
    k_zero<<<1, 128, 0, stream>>>(colsum, DD);
    k_zero<<<1, 128, 0, stream>>>(colsq, DD);
    k_bn_stats<<<256, 128, 0, stream>>>(agg, colsum, colsq, N);
    k_bn_finalize<<<1, 128, 0, stream>>>(colsum, colsq, gamma + (size_t)l * DD,
                                         beta + (size_t)l * DD, scale, shift, N);
    k_bn_relu_cast<<<CDIV(N * DD, 256), 256, 0, stream>>>(agg, scale, shift, xcur, xb, N * DD);
  }

  // global mean pool per graph
  k_zero<<<CDIV(G * DD, 256), 256, 0, stream>>>(gsum, G * DD);
  k_zero<<<CDIV(G, 256), 256, 0, stream>>>(cnt, G);
  k_pool_cnt<<<CDIV(N, 256), 256, 0, stream>>>(batch, cnt, N);
  k_pool_sum<<<CDIV(N * 32, 256), 256, 0, stream>>>(batch, xcur, gsum, N);
  k_pool_div_cast<<<CDIV(G * DD, 256), 256, 0, stream>>>(gsum, cnt, gbf, G * DD);

  // MLP head: relu(g @ W1 + b1) @ W2 + b2
  k_pack_b<<<CDIV(HH * 64, 256), 256, 0, stream>>>(W1, wbp, HH);
  int hwaves = (G / 16) * (HH / 128);
  k_gemm_bf16_wmma<<<CDIV(hwaves, 8), 256, 0, stream>>>(gbf, wbp, hidden, G, HH);
  k_bias_relu<<<CDIV(G * HH, 256), 256, 0, stream>>>(hidden, b1, G * HH);
  k_head2<<<CDIV(G * 32, 256), 256, 0, stream>>>(hidden, W2, b2, out, G);
}
